// SentenceLevelEncoder_24137716203568
// MI455X (gfx1250) — matrix operand order repaired and checked by reading, not compiled, not run on hardware
//
#include <hip/hip_runtime.h>
#include <hip/hip_bf16.h>

typedef __attribute__((ext_vector_type(16))) __bf16        v16bf;
typedef __attribute__((ext_vector_type(8)))  float         v8f;
typedef __attribute__((ext_vector_type(8)))  unsigned short u16x8;
typedef __attribute__((ext_vector_type(4)))  float         f32x4;
typedef __attribute__((ext_vector_type(4)))  __bf16        bf16x4;

#define B_   32
#define T_   32
#define O_   64
#define H_   512
#define HL_  256
#define INF_ 100000.0f

// ---------------------------------------------------------------------------
// CDNA5 async copy: 16 bytes per lane, global -> LDS, tracked by ASYNCcnt.
// ---------------------------------------------------------------------------
__device__ __forceinline__ void async_copy16(const __bf16* g, __bf16* l) {
  unsigned lds_off = (unsigned)(unsigned long long)l;   // low 32b = LDS offset
  unsigned long long ga = (unsigned long long)g;
  asm volatile("global_load_async_to_lds_b128 %0, %1, off"
               :: "v"(lds_off), "v"(ga) : "memory");
}

template <int N>
__device__ __forceinline__ void wait_async() {
  asm volatile("s_wait_asynccnt %0" :: "i"(N) : "memory");
}

__device__ __forceinline__ v16bf frag_from(u16x8 lo, u16x8 hi) {
  union { u16x8 h[2]; v16bf v; } u;
  u.h[0] = lo; u.h[1] = hi;
  return u.v;
}

// ---------------------------------------------------------------------------
// f32 -> bf16 bulk convert (4-wide)
// ---------------------------------------------------------------------------
__global__ void cvt_bf16_kernel(const f32x4* __restrict__ src,
                                bf16x4* __restrict__ dst, int n4) {
  int i = blockIdx.x * blockDim.x + threadIdx.x;
  if (i < n4) {
    f32x4 v = src[i];
    bf16x4 o;
    o[0] = (__bf16)v[0]; o[1] = (__bf16)v[1];
    o[2] = (__bf16)v[2]; o[3] = (__bf16)v[3];
    dst[i] = o;
  }
}

// ---------------------------------------------------------------------------
// bf16 GEMM: Y = X(bf16) @ W(bf16,N x K)^T + bias(+bias2)
// One wave per 16x16 tile, 8 waves/block. Per-wave double-buffered LDS tiles
// filled by GLOBAL_LOAD_ASYNC_TO_LDS_B128; fragments read back as b128 DS ops.
// Optional dual f32 / bf16 outputs.
// ---------------------------------------------------------------------------
__global__ void gemm_bf16_kernel(const __bf16* __restrict__ X, int ldX,
                                 const __bf16* __restrict__ W,
                                 const float* __restrict__ bias,
                                 const float* __restrict__ bias2,
                                 float* __restrict__ Y, int ldY,
                                 __bf16* __restrict__ Yb, int ldYb,
                                 int M, int N, int K) {
  __shared__ __bf16 sA[8][2][16 * 32];   // per-wave double buffer: 16 KB
  __shared__ __bf16 sB[8][2][16 * 32];   // 16 KB

  int lane = threadIdx.x & 31;
  int wave = threadIdx.x >> 5;
  int ntN = N >> 4;
  int tiles = (M >> 4) * ntN;
  int tile = blockIdx.x * 8 + wave;
  if (tile >= tiles) return;
  int m0 = (tile / ntN) << 4;
  int n0 = (tile % ntN) << 4;
  int half = lane >> 4;
  int l15 = lane & 15;

  const __bf16* xrow = X + (m0 + l15) * ldX + half * 16;  // lane's 32B slice
  const __bf16* wrow = W + (n0 + l15) * K + half * 16;

  int nk = K >> 5;

  // stage k-block 0 into buffer 0
  {
    __bf16* la = &sA[wave][0][l15 * 32 + half * 16];
    __bf16* lb = &sB[wave][0][l15 * 32 + half * 16];
    async_copy16(xrow, la);      async_copy16(xrow + 8, la + 8);
    async_copy16(wrow, lb);      async_copy16(wrow + 8, lb + 8);
  }

  v8f acc = {};
  for (int i = 0; i < nk; ++i) {
    int p = i & 1;
    if (i + 1 < nk) {  // prefetch next k-block into the other buffer
      int k = (i + 1) << 5;
      __bf16* la = &sA[wave][p ^ 1][l15 * 32 + half * 16];
      __bf16* lb = &sB[wave][p ^ 1][l15 * 32 + half * 16];
      async_copy16(xrow + k, la);  async_copy16(xrow + k + 8, la + 8);
      async_copy16(wrow + k, lb);  async_copy16(wrow + k + 8, lb + 8);
      wait_async<4>();             // wait for the OLDER stage (4 newest pending)
    } else {
      wait_async<0>();
    }
    const u16x8* pa = (const u16x8*)&sA[wave][p][0];
    const u16x8* pb = (const u16x8*)&sB[wave][p][0];
    v16bf a = frag_from(pa[l15 * 4 + half * 2], pa[l15 * 4 + half * 2 + 1]);
    v16bf b = frag_from(pb[l15 * 4 + half * 2], pb[l15 * 4 + half * 2 + 1]);
    acc = __builtin_amdgcn_wmma_f32_16x16x32_bf16(
        false, a, false, b, (short)0, acc, false, false);
  }

  float bb = bias ? bias[n0 + l15] : 0.0f;
  if (bias2) bb += bias2[n0 + l15];
#pragma unroll
  for (int r = 0; r < 8; ++r) {
    int m = m0 + r + (half << 3);
    float v = acc[r] + bb;
    if (Y)  Y[m * ldY + n0 + l15] = v;
    if (Yb) Yb[m * ldYb + n0 + l15] = (__bf16)v;
  }
}

// ---------------------------------------------------------------------------
// Fused additive attention for one (b,t); writes bf16 directly into feats.
// ---------------------------------------------------------------------------
__global__ void attn_kernel(const float* __restrict__ Wf,   // (B*T, H)
                            const float* __restrict__ U,    // (B, L, H)
                            const float* __restrict__ bvec, // (H)
                            const float* __restrict__ wv,   // (H)
                            const float* __restrict__ wb,   // (1)
                            const int*   __restrict__ mask, // (B, L) or null
                            __bf16* __restrict__ outb,      // row stride ldOut
                            int ldOut, int colOff, int L) {
  __shared__ float sc[O_];
  int bt = blockIdx.x;
  int b = bt / T_;
  int tid = threadIdx.x;
  int lane = tid & 31, wave = tid >> 5;
  const float* wf = Wf + bt * H_;

  int perWave = L >> 3;
  for (int j = 0; j < perWave; ++j) {
    int o = wave * perWave + j;
    const float* u = U + (b * L + o) * H_;
    float p = 0.f;
    for (int h = lane; h < H_; h += 32) {
      float afv = wf[h] + u[h] + bvec[h];
      p += tanhf(afv) * wv[h];
    }
#pragma unroll
    for (int off = 16; off > 0; off >>= 1) p += __shfl_xor(p, off);
    if (lane == 0) sc[o] = p;
  }
  __syncthreads();

  if (tid == 0) {
    float wbv = wb[0];
    float mx = -3.4e38f;
    for (int o = 0; o < L; ++o) {
      float s = sc[o] + wbv;
      if (mask) s -= (float)mask[b * L + o] * INF_;
      sc[o] = s;
      mx = fmaxf(mx, s);
    }
    float sum = 0.f;
    for (int o = 0; o < L; ++o) { sc[o] = expf(sc[o] - mx); sum += sc[o]; }
    float inv = 1.f / sum;
    for (int o = 0; o < L; ++o) sc[o] *= inv;
  }
  __syncthreads();

  for (int h = tid; h < H_; h += blockDim.x) {
    float base = wf[h] + bvec[h];
    float accv = 0.f;
    for (int o = 0; o < L; ++o)
      accv += sc[o] * (base + U[(b * L + o) * H_ + h]);
    outb[bt * ldOut + colOff + h] = (__bf16)accv;
  }
}

// ---------------------------------------------------------------------------
// Persistent LSTM scan: one 1024-thread workgroup, state in LDS.
// h fragments: b128 DS loads; whh (pre-converted bf16): b128 global loads.
// ---------------------------------------------------------------------------
__global__ void lstm_scan_kernel(const float* __restrict__ xg,    // (B*T,1024)
                                 const __bf16* __restrict__ whh,  // (1024,256)
                                 float* __restrict__ out,         // (B*T,512)
                                 int colOff, int reverse) {
  extern __shared__ unsigned char smem[];
  float*  g_buf = (float*)smem;                                 // 32*1024 f32
  float*  c_lds = (float*)(smem + 32 * 1024 * 4);               // 32*256 f32
  __bf16* h_bf  = (__bf16*)(smem + (32 * 1024 + 32 * 256) * 4); // 32*256 bf16

  int tid = threadIdx.x;
  int lane = tid & 31;
  int wave = tid >> 5;
  int half = lane >> 4;
  int l15 = lane & 15;

  for (int i = tid; i < B_ * HL_; i += 1024) {
    c_lds[i] = 0.f;
    h_bf[i] = (__bf16)0.f;
  }
  __syncthreads();

  for (int s = 0; s < T_; ++s) {
    int t = reverse ? (T_ - 1 - s) : s;

#pragma unroll
    for (int j = 0; j < 4; ++j) {
      int tile = wave * 4 + j;            // 2 Mtiles x 64 Ntiles
      int m0 = (tile >> 6) << 4;
      int n0 = (tile & 63) << 4;
      v8f acc = {};
      const u16x8* hrow = (const u16x8*)(h_bf + (m0 + l15) * HL_);
      const u16x8* wrow = (const u16x8*)(whh + (n0 + l15) * HL_);
#pragma unroll
      for (int k0 = 0; k0 < HL_; k0 += 32) {
        int c = k0 >> 3;                  // u16x8 chunk index
        v16bf a = frag_from(hrow[c + half * 2], hrow[c + half * 2 + 1]);
        v16bf b = frag_from(wrow[c + half * 2], wrow[c + half * 2 + 1]);
        acc = __builtin_amdgcn_wmma_f32_16x16x32_bf16(
            false, a, false, b, (short)0, acc, false, false);
      }
#pragma unroll
      for (int r = 0; r < 8; ++r) {
        int m = m0 + r + (half << 3);
        int n = n0 + l15;
        g_buf[m * 1024 + n] = acc[r] + xg[(m * T_ + t) * 1024 + n];
      }
    }
    __syncthreads();

#pragma unroll
    for (int i = 0; i < 8; ++i) {
      int idx = tid + (i << 10);
      int b = idx >> 8;
      int hl = idx & 255;
      float iv = g_buf[b * 1024 + hl];
      float fv = g_buf[b * 1024 + 256 + hl];
      float gv = g_buf[b * 1024 + 512 + hl];
      float ov = g_buf[b * 1024 + 768 + hl];
      iv = 1.f / (1.f + expf(-iv));
      fv = 1.f / (1.f + expf(-fv));
      gv = tanhf(gv);
      ov = 1.f / (1.f + expf(-ov));
      float c = fv * c_lds[idx] + iv * gv;
      float h = ov * tanhf(c);
      c_lds[idx] = c;
      h_bf[idx] = (__bf16)h;
      out[(b * T_ + t) * H_ + colOff + hl] = h;
    }
    __syncthreads();
  }
}

__global__ void rowmax_kernel(const float* __restrict__ out,
                              float* __restrict__ video,
                              __bf16* __restrict__ video_bf) {
  int idx = blockIdx.x * blockDim.x + threadIdx.x;
  if (idx >= B_ * H_) return;
  int b = idx / H_, h = idx % H_;
  float m = -3.4e38f;
  for (int t = 0; t < T_; ++t) m = fmaxf(m, out[(b * T_ + t) * H_ + h]);
  video[idx] = m;
  video_bf[idx] = (__bf16)m;
}

extern "C" void kernel_launch(void* const* d_in, const int* in_sizes, int n_in,
                              void* d_out, int out_size, void* d_ws, size_t ws_size,
                              hipStream_t stream) {
  const float* feature2ds      = (const float*)d_in[0];
  const float* vp_features     = (const float*)d_in[1];
  const float* object_features = (const float*)d_in[2];
  const int*   objects_mask    = (const int*)d_in[3];
  const float* w2d  = (const float*)d_in[4];
  const float* b2d  = (const float*)d_in[5];
  const float* Ww   = (const float*)d_in[6];
  const float* Wb   = (const float*)d_in[7];
  const float* Uow  = (const float*)d_in[8];
  const float* Uob  = (const float*)d_in[9];
  const float* Umw  = (const float*)d_in[10];
  const float* Umb  = (const float*)d_in[11];
  const float* bo   = (const float*)d_in[12];
  const float* bm   = (const float*)d_in[13];
  const float* wow  = (const float*)d_in[14];
  const float* wob  = (const float*)d_in[15];
  const float* wmw  = (const float*)d_in[16];
  const float* wmb  = (const float*)d_in[17];
  const float* w_ih_f = (const float*)d_in[18];
  const float* w_hh_f = (const float*)d_in[19];
  const float* b_ih_f = (const float*)d_in[20];
  const float* b_hh_f = (const float*)d_in[21];
  const float* w_ih_b = (const float*)d_in[22];
  const float* w_hh_b = (const float*)d_in[23];
  const float* b_ih_b = (const float*)d_in[24];
  const float* b_hh_b = (const float*)d_in[25];
  const float* fcw  = (const float*)d_in[26];
  const float* fcb  = (const float*)d_in[27];

  // ---- workspace carve-up ----
  char* p = (char*)d_ws;
  auto allocF = [&](size_t n) { float* r = (float*)p; p += n * 4; return r; };
  auto allocH = [&](size_t n) { __bf16* r = (__bf16*)p; p += n * 2; return r; };

  float* Wf    = allocF(1024 * 512);
  float* Uo    = allocF(2048 * 512);
  float* Um    = allocF(1024 * 512);
  float* xg_f  = allocF(1024 * 1024);
  float* xg_b  = allocF(1024 * 1024);
  float* video = allocF(32 * 512);

  __bf16* feats_bf = allocH(1024 * 1536);  // [f2d | attn_mot | attn_obj]
  __bf16* x2d_bf   = allocH(1024 * 2048);
  __bf16* obj_bf   = allocH(2048 * 512);
  __bf16* vp_bf    = allocH(1024 * 512);
  __bf16* w2d_bf   = allocH(512 * 2048);
  __bf16* Ww_bf    = allocH(512 * 512);
  __bf16* Uow_bf   = allocH(512 * 512);
  __bf16* Umw_bf   = allocH(512 * 512);
  __bf16* wihf_bf  = allocH(1024 * 1536);
  __bf16* wihb_bf  = allocH(1024 * 1536);
  __bf16* whhf_bf  = allocH(1024 * 256);
  __bf16* whhb_bf  = allocH(1024 * 256);
  __bf16* fcw_bf   = allocH(768 * 512);
  __bf16* video_bf = allocH(32 * 512);

  float* outp = (float*)d_out;
  float* vidp = outp + 1024 * 512;

  auto cvt = [&](const float* s, __bf16* d, int n) {
    int n4 = n >> 2;
    cvt_bf16_kernel<<<(n4 + 255) / 256, 256, 0, stream>>>(
        (const f32x4*)s, (bf16x4*)d, n4);
  };
  auto grid = [](int M, int N) { return ((M / 16) * (N / 16) + 7) / 8; };

  // ---- one-time f32 -> bf16 conversion of inputs & weights ----
  cvt(feature2ds, x2d_bf, 1024 * 2048);
  cvt(object_features, obj_bf, 2048 * 512);
  cvt(vp_features, vp_bf, 1024 * 512);
  cvt(w2d, w2d_bf, 512 * 2048);
  cvt(Ww, Ww_bf, 512 * 512);
  cvt(Uow, Uow_bf, 512 * 512);
  cvt(Umw, Umw_bf, 512 * 512);
  cvt(w_ih_f, wihf_bf, 1024 * 1536);
  cvt(w_ih_b, wihb_bf, 1024 * 1536);
  cvt(w_hh_f, whhf_bf, 1024 * 256);
  cvt(w_hh_b, whhb_bf, 1024 * 256);
  cvt(fcw, fcw_bf, 768 * 512);

  // f2d -> feats_bf[:, 0:512] (bf16 only)
  gemm_bf16_kernel<<<grid(1024, 512), 256, 0, stream>>>(
      x2d_bf, 2048, w2d_bf, b2d, nullptr,
      nullptr, 0, feats_bf, 1536, 1024, 512, 2048);
  // Wf = f2d @ Ww^T + Wb (f32)
  gemm_bf16_kernel<<<grid(1024, 512), 256, 0, stream>>>(
      feats_bf, 1536, Ww_bf, Wb, nullptr,
      Wf, 512, nullptr, 0, 1024, 512, 512);
  // Uo, Um (f32)
  gemm_bf16_kernel<<<grid(2048, 512), 256, 0, stream>>>(
      obj_bf, 512, Uow_bf, Uob, nullptr,
      Uo, 512, nullptr, 0, 2048, 512, 512);
  gemm_bf16_kernel<<<grid(1024, 512), 256, 0, stream>>>(
      vp_bf, 512, Umw_bf, Umb, nullptr,
      Um, 512, nullptr, 0, 1024, 512, 512);

  // attention -> feats_bf columns 512.. / 1024..
  attn_kernel<<<1024, 256, 0, stream>>>(Wf, Um, bm, wmw, wmb, nullptr,
                                        feats_bf, 1536, 512, 32);
  attn_kernel<<<1024, 256, 0, stream>>>(Wf, Uo, bo, wow, wob, objects_mask,
                                        feats_bf, 1536, 1024, 64);

  // LSTM input GEMMs (bias = b_ih + b_hh)
  gemm_bf16_kernel<<<grid(1024, 1024), 256, 0, stream>>>(
      feats_bf, 1536, wihf_bf, b_ih_f, b_hh_f,
      xg_f, 1024, nullptr, 0, 1024, 1024, 1536);
  gemm_bf16_kernel<<<grid(1024, 1024), 256, 0, stream>>>(
      feats_bf, 1536, wihb_bf, b_ih_b, b_hh_b,
      xg_b, 1024, nullptr, 0, 1024, 1024, 1536);

  // LSTM scans
  size_t lstm_lds = (size_t)(32 * 1024 + 32 * 256) * 4 + (size_t)(32 * 256) * 2;
  lstm_scan_kernel<<<1, 1024, lstm_lds, stream>>>(xg_f, whhf_bf, outp, 0, 0);
  lstm_scan_kernel<<<1, 1024, lstm_lds, stream>>>(xg_b, whhb_bf, outp, 256, 1);

  // video = max over T, then FC
  rowmax_kernel<<<(B_ * H_ + 255) / 256, 256, 0, stream>>>(outp, video, video_bf);
  gemm_bf16_kernel<<<grid(32, 768), 256, 0, stream>>>(
      video_bf, 512, fcw_bf, fcb, nullptr,
      vidp, 768, nullptr, 0, 32, 768, 512);
}
